// CrossMambaBlock_88699664597225
// MI455X (gfx1250) — matrix operand order repaired and checked by reading, not compiled, not run on hardware
//
#include <hip/hip_runtime.h>
#include <math.h>

// ---------------------------------------------------------------------------
// CrossMambaBlock for MI455X (gfx1250, wave32).
//
// KEY INSIGHT: the reference uses x.reshape(b*h, w, c) / x.reshape(b*w, h, c),
// which are FLAT reshapes (not transposes) of the (4,256,64,64) buffer. Both
// views are the identical contiguous (16384 x 256) row-major matrix, and the
// output reshape is flat-preserving too. So every GEMM is a plain contiguous
// GEMM (no gather/scatter); the two directions differ only in parameters.
//
// The three big GEMMs run on matrix cores via v_wmma_f32_16x16x32_bf16
// (bf16 in / f32 accumulate), tiles staged through double-buffered LDS with
// branchless f32->bf16 conversion; global loads of tile k+1 overlap the
// WMMAs of tile k. Conv + selective scan are wave32 VALU kernels (the scan is
// the sequential-over-L part; 256x512 independent (seq,channel) lanes with 16
// h-states in registers give full occupancy). The whole problem is memory
// bound (~30 GFLOP vs ~400MB of intermediate traffic @ 23.3 TB/s; working set
// fits in the 192MB L2), so coalesced streaming + fused scan is the win.
// ---------------------------------------------------------------------------

#define D_MODEL  256
#define D_INNER  512
#define D_STATE  16
#define DT_RANK  16
#define SEQ_LEN  64
#define NSEQ     256                 // 16384 / 64
#define MTOT     (NSEQ * SEQ_LEN)    // 16384

typedef __attribute__((ext_vector_type(16))) __bf16         v16bf;
typedef __attribute__((ext_vector_type(8)))  float          v8f;
typedef __attribute__((ext_vector_type(4)))  unsigned int   uint4v;
typedef __attribute__((ext_vector_type(4)))  unsigned short ushort4v;
typedef __attribute__((ext_vector_type(4)))  float          float4v;

union FragU { uint4v u4[2]; v16bf bf; };

// Branchless RNE f32->bf16 (NaN select lowers to v_cndmask, no EXEC branches).
static __device__ __forceinline__ unsigned short f2bf(float f) {
    unsigned int u = __float_as_uint(f);
    unsigned int rne = (u + 0x7fffu + ((u >> 16) & 1u)) >> 16;
    unsigned int nan = (u >> 16) | 0x0040u;
    bool is_nan = (u & 0x7fffffffu) > 0x7f800000u;
    return (unsigned short)(is_nan ? nan : rne);
}

static __device__ __forceinline__ float silu(float v) {
    return v / (1.0f + __expf(-v));
}

// ------------------------- weight conversion kernels -----------------------

__global__ __launch_bounds__(256) void f32_to_bf16_kernel(
    const float* __restrict__ src, unsigned short* __restrict__ dst, int n) {
    int i = blockIdx.x * 256 + threadIdx.x;
    if (i < n) dst[i] = f2bf(src[i]);
}

// W_x is (48, 512) -> pad to (64, 512) bf16, zero rows 48..63
__global__ __launch_bounds__(256) void pad_wx_kernel(
    const float* __restrict__ wx, unsigned short* __restrict__ dst) {
    int i = blockIdx.x * 256 + threadIdx.x;   // 64*512 = 32768
    int row = i >> 9;
    dst[i] = (row < (DT_RANK + 2 * D_STATE)) ? f2bf(wx[i]) : (unsigned short)0;
}

// ------------------------------ WMMA GEMM ----------------------------------
// D[m,n] = sum_k A[m*lda + k] * W[n*K + k]   (W row-major [N,K] bf16)
// c_mode: 0 = Out[m*ldo + n]  = val
//         1 = Out[m*ldo + n]  = Xres[m*ldo + n] + val      (dir 1 epilogue)
//         2 = Out[m*ldo + n] += val                        (dir 2 epilogue)

#define BM 128
#define BN 64
#define BK 32
#define LDSP (BK + 8)   // 40 ushorts (80B) per row: 16B-aligned rows, staggered banks

__global__ __launch_bounds__(256) void wmma_gemm_kernel(
    const float* __restrict__ A, const unsigned short* __restrict__ Wbf,
    float* __restrict__ Out, const float* __restrict__ Xres,
    int K, int lda, int ldo, int c_mode)
{
    // double-buffered tiles: ~31KB of the 320KB WGP LDS pool
    __shared__ __attribute__((aligned(16))) unsigned short As[2][BM][LDSP];
    __shared__ __attribute__((aligned(16))) unsigned short Bs[2][BN][LDSP];

    const int tid   = threadIdx.x;
    const int lane  = tid & 31;
    const int wave  = tid >> 5;
    const int wm    = wave & 3;     // 4 waves tile M (32 rows each)
    const int wn    = wave >> 2;    // 2 waves tile N (32 cols each)
    const int m0    = blockIdx.x * BM;
    const int n0    = blockIdx.y * BN;
    const int l16   = lane & 15;
    const int khalf = lane >> 4;

    v8f acc[2][2];
    for (int mi = 0; mi < 2; ++mi)
        for (int ni = 0; ni < 2; ++ni)
            for (int e = 0; e < 8; ++e) acc[mi][ni][e] = 0.0f;

    // tile stager: float4 global loads -> 4x branchless f2bf -> b64 LDS store
    auto stage = [&](int k0, int bi) {
#pragma unroll
        for (int it = 0; it < (BM * BK) / 4 / 256; ++it) {   // 4 iters
            int q = it * 256 + tid;
            int r = q >> 3, c = (q & 7) * 4;
            float4v f = *(const float4v*)(A + (size_t)(m0 + r) * lda + k0 + c);
            ushort4v hv;
            hv[0] = f2bf(f[0]); hv[1] = f2bf(f[1]);
            hv[2] = f2bf(f[2]); hv[3] = f2bf(f[3]);
            *(ushort4v*)(&As[bi][r][c]) = hv;
        }
#pragma unroll
        for (int it = 0; it < (BN * BK) / 4 / 256; ++it) {   // 2 iters
            int q = it * 256 + tid;
            int r = q >> 3, c = (q & 7) * 4;
            *(ushort4v*)(&Bs[bi][r][c]) =
                *(const ushort4v*)(Wbf + (size_t)(n0 + r) * K + k0 + c);
        }
    };

    // fragment build + 4 WMMAs from buffer bi (constant at each call site)
    auto compute = [&](int bi) {
        // ISA 16-bit layouts (wave32):
        //  A 16x32: lane%16 = M; elems 0..7  -> K = khalf*8 + (0..7)
        //                        elems 8..15 -> K = 16 + khalf*8 + (0..7)
        //  B 32x16: lane%16 = N; same K mapping as A (khalf*8 / 16 + khalf*8)
        FragU afrag[2], bfrag[2];
#pragma unroll
        for (int mi = 0; mi < 2; ++mi) {
            const unsigned short* arow = &As[bi][wm * 32 + mi * 16 + l16][0];
            afrag[mi].u4[0] = *(const uint4v*)(arow + khalf * 8);
            afrag[mi].u4[1] = *(const uint4v*)(arow + 16 + khalf * 8);
        }
#pragma unroll
        for (int ni = 0; ni < 2; ++ni) {
            const unsigned short* brow = &Bs[bi][wn * 32 + ni * 16 + l16][0];
            bfrag[ni].u4[0] = *(const uint4v*)(brow + khalf * 8);
            bfrag[ni].u4[1] = *(const uint4v*)(brow + 16 + khalf * 8);
        }
#pragma unroll
        for (int mi = 0; mi < 2; ++mi)
#pragma unroll
            for (int ni = 0; ni < 2; ++ni)
                acc[mi][ni] = __builtin_amdgcn_wmma_f32_16x16x32_bf16(
                    false, afrag[mi].bf, false, bfrag[ni].bf,
                    (short)0, acc[mi][ni], false, false);
    };

    const int nK = K / BK;   // always even here (8 or 16)
    stage(0, 0);

    for (int kt = 0; kt < nK; kt += 2) {
        __syncthreads();
        if (kt + 1 < nK) stage((kt + 1) * BK, 1);
        compute(0);
        __syncthreads();
        if (kt + 2 < nK) stage((kt + 2) * BK, 0);
        compute(1);
    }

    // Epilogue. C/D layout: VGPR v -> M = v + 8*(lane/16), N = lane%16.
    // n = lane%16 is the fastest-varying store index -> fully coalesced.
#pragma unroll
    for (int mi = 0; mi < 2; ++mi)
#pragma unroll
        for (int ni = 0; ni < 2; ++ni)
#pragma unroll
            for (int v = 0; v < 8; ++v) {
                int m = m0 + wm * 32 + mi * 16 + khalf * 8 + v;
                int n = n0 + wn * 32 + ni * 16 + l16;
                size_t o = (size_t)m * ldo + n;
                float val = acc[mi][ni][v];
                if (c_mode == 0)       Out[o] = val;
                else if (c_mode == 1)  Out[o] = Xres[o] + val;
                else                   Out[o] += val;
            }
}

// -------------------- depthwise causal conv (taps=4) + SiLU ----------------
// in: first half of xz, rows m = seq*64 + l, row stride 1024; out: xc (MTOT x 512)
__global__ __launch_bounds__(256) void conv_silu_kernel(
    const float* __restrict__ xz, const float* __restrict__ convw,
    const float* __restrict__ convb, float* __restrict__ xc)
{
    int g = blockIdx.x * 256 + threadIdx.x;     // 256*512 = 131072 threads
    int seq = g >> 9;
    int d   = g & 511;
    float w0 = convw[d * 4 + 0], w1 = convw[d * 4 + 1];
    float w2 = convw[d * 4 + 2], w3 = convw[d * 4 + 3];
    float bb = convb[d];
    float xm3 = 0.f, xm2 = 0.f, xm1 = 0.f;
    for (int l = 0; l < SEQ_LEN; ++l) {
        size_t m = (size_t)seq * SEQ_LEN + l;
        float xl = xz[m * 1024 + d];
        float s  = xm3 * w0 + xm2 * w1 + xm1 * w2 + xl * w3 + bb;
        xc[m * 512 + d] = silu(s);
        xm3 = xm2; xm2 = xm1; xm1 = xl;
    }
}

// ----------------------------- selective scan ------------------------------
// One block per sequence (256 blocks), 512 threads = 16 wave32s, thread = d.
// Per step l, threads 0..47 broadcast dbc[b,l,0:48] through LDS (dt-proj,B,C).
// Fuses: dt = softplus(dbc[:,:16] @ W_dt^T + b_dt), h-state recurrence,
// D_skip residual, and the SiLU(z) gate.
__global__ __launch_bounds__(512) void scan_kernel(
    const float* __restrict__ dbc, const float* __restrict__ xc,
    const float* __restrict__ xz, const float* __restrict__ Wdt,
    const float* __restrict__ bdt, const float* __restrict__ Alog,
    const float* __restrict__ Dskip, float* __restrict__ ybuf)
{
    __shared__ float sd[2][48];
    const int b = blockIdx.x;
    const int d = threadIdx.x;

    float wdt[DT_RANK];
#pragma unroll
    for (int r = 0; r < DT_RANK; ++r) wdt[r] = Wdt[d * DT_RANK + r];
    float Av[D_STATE];
#pragma unroll
    for (int n = 0; n < D_STATE; ++n) Av[n] = -__expf(Alog[d * D_STATE + n]);
    float bd = bdt[d];
    float ds = Dskip[d];
    float h[D_STATE];
#pragma unroll
    for (int n = 0; n < D_STATE; ++n) h[n] = 0.f;

    for (int l = 0; l < SEQ_LEN; ++l) {
        int buf = l & 1;
        if (threadIdx.x < 48)
            sd[buf][threadIdx.x] = dbc[((size_t)b * SEQ_LEN + l) * 64 + threadIdx.x];
        __syncthreads();

        float dtv = bd;
#pragma unroll
        for (int r = 0; r < DT_RANK; ++r) dtv += sd[buf][r] * wdt[r];
        dtv = (dtv > 20.f) ? dtv : log1pf(__expf(dtv));   // softplus

        size_t m = (size_t)b * SEQ_LEN + l;
        float xv  = xc[m * 512 + d];
        float dtx = dtv * xv;
        float accv = 0.f;
#pragma unroll
        for (int n = 0; n < D_STATE; ++n) {
            float dA = __expf(dtv * Av[n]);
            h[n] = dA * h[n] + dtx * sd[buf][16 + n];
            accv += h[n] * sd[buf][32 + n];
        }
        float zv = xz[m * 1024 + 512 + d];
        ybuf[m * 512 + d] = (accv + xv * ds) * silu(zv);
    }
}

// ------------------------------- launcher ----------------------------------

extern "C" void kernel_launch(void* const* d_in, const int* in_sizes, int n_in,
                              void* d_out, int out_size, void* d_ws, size_t ws_size,
                              hipStream_t stream) {
    // x.reshape(b*h,w,c) and x.reshape(b*w,h,c) are flat reshapes of the same
    // buffer -> both mamba passes consume x as a contiguous (16384 x 256) matrix.
    const float* x = (const float*)d_in[0];
    float* out = (float*)d_out;

    // workspace layout (f32 unless noted); reused across the two directions
    float* xz   = (float*)d_ws;                       // 16384 x 1024
    float* xc   = xz   + (size_t)MTOT * 1024;         // 16384 x 512
    float* dbc  = xc   + (size_t)MTOT * 512;          // 16384 x 64 (48 used)
    float* ybuf = dbc  + (size_t)MTOT * 64;           // 16384 x 512
    unsigned short* win_bf  = (unsigned short*)(ybuf + (size_t)MTOT * 512); // 1024x256
    unsigned short* wx_bf   = win_bf + 1024 * 256;    // 64x512 (padded)
    unsigned short* wout_bf = wx_bf  + 64 * 512;      // 256x512

    for (int dir = 0; dir < 2; ++dir) {
        int base = 1 + dir * 9;
        const float* W_in   = (const float*)d_in[base + 0];
        const float* convw  = (const float*)d_in[base + 1];
        const float* convb  = (const float*)d_in[base + 2];
        const float* W_x    = (const float*)d_in[base + 3];
        const float* W_dt   = (const float*)d_in[base + 4];
        const float* b_dt   = (const float*)d_in[base + 5];
        const float* A_log  = (const float*)d_in[base + 6];
        const float* D_skip = (const float*)d_in[base + 7];
        const float* W_out  = (const float*)d_in[base + 8];

        f32_to_bf16_kernel<<<dim3(1024), dim3(256), 0, stream>>>(W_in, win_bf, 1024 * 256);
        pad_wx_kernel<<<dim3(128), dim3(256), 0, stream>>>(W_x, wx_bf);
        f32_to_bf16_kernel<<<dim3(512), dim3(256), 0, stream>>>(W_out, wout_bf, 256 * 512);

        // GEMM1: xz = x @ W_in^T   (M=16384, N=1024, K=256), contiguous A
        wmma_gemm_kernel<<<dim3(MTOT / BM, 1024 / BN), dim3(256), 0, stream>>>(
            x, win_bf, xz, nullptr, 256, 256, 1024, 0);

        // depthwise conv + SiLU -> xc
        conv_silu_kernel<<<dim3((NSEQ * D_INNER) / 256), dim3(256), 0, stream>>>(
            xz, convw, convb, xc);

        // GEMM2: dbc = xc @ W_x^T  (M=16384, N=64 padded, K=512)
        wmma_gemm_kernel<<<dim3(MTOT / BM, 64 / BN), dim3(256), 0, stream>>>(
            xc, wx_bf, dbc, nullptr, 512, 512, 64, 0);

        // selective scan (fuses dt-proj + softplus + gate) -> ybuf
        scan_kernel<<<dim3(NSEQ), dim3(512), 0, stream>>>(
            dbc, xc, xz, W_dt, b_dt, A_log, D_skip, ybuf);

        // GEMM3: out = x + y1 (dir 0), then out += y2 (dir 1); contiguous epilogue
        wmma_gemm_kernel<<<dim3(MTOT / BM, 256 / BN), dim3(256), 0, stream>>>(
            ybuf, wout_bf, out, x, 512, 512, 256, 1 + dir);
    }
}
